// Policy_55413668053172
// MI455X (gfx1250) — matrix operand order repaired and checked by reading, not compiled, not run on hardware
//
#include <hip/hip_runtime.h>
#include <hip/hip_bf16.h>
#include <math.h>

typedef __attribute__((ext_vector_type(16))) _Float16 v16h;
typedef __attribute__((ext_vector_type(8)))  _Float16 v8h;
typedef __attribute__((ext_vector_type(8)))  float    v8f;
typedef _Float16 f16;

#define LDSP 136   // padded LDS row stride (halves) for 16x128 tiles

// NOTE: every GEMM row-count R in this file is a multiple of 16 and grids are
// exactly R/16, so tile row guards are omitted throughout.

// One-time: Wt[n*128+k] = (f16) W[k*128+n]   (128x128)
__global__ void transpose_w_kernel(const float* __restrict__ W, f16* __restrict__ Wt) {
  const int o = blockIdx.x * 256 + threadIdx.x;     // grid 64 x 256
  const int n = o >> 7, k = o & 127;
  Wt[o] = (f16)W[k * 128 + n];
}

// ---------------------------------------------------------------------------
// Wave-level 16x16 output tile of (16 x 128) @ (128 x 128), f16 in, f32 acc.
// sX: LDS, 16 rows x LDSP halves. Wt: f16 column-major (n-major) 128x128.
// A frag: lane holds row lane&15; element j -> K = kb + (j&7) + 16*(j>>3) + 8*hs
//   == two contiguous 8-half runs at kb+8*hs and kb+16+8*hs.
// B frag: lane holds col n0+(lane&15); element j -> K = kb + (j&7) + 16*(j>>3) + 8*hs
//   == two contiguous 8-half runs in Wt row n (same as A).
// ---------------------------------------------------------------------------
__device__ __forceinline__ v8f tile_gemm_128(const f16* sX, const f16* __restrict__ Wt,
                                             int n0, int lane) {
  v8f acc = {};
  const int hs   = lane >> 4;
  const int lrow = lane & 15;
  const f16* xrow = sX + lrow * LDSP;
  const f16* wrow = Wt + (n0 + lrow) * 128;
  #pragma unroll
  for (int kt = 0; kt < 4; ++kt) {
    const int kb = kt * 32;
    const v8h alo = *(const v8h*)(xrow + kb + (hs << 3));
    const v8h ahi = *(const v8h*)(xrow + kb + 16 + (hs << 3));
    const v8h blo = *(const v8h*)(wrow + kb + (hs << 3));
    const v8h bhi = *(const v8h*)(wrow + kb + 16 + (hs << 3));
    const v16h a = __builtin_shufflevector(alo, ahi, 0,1,2,3,4,5,6,7,8,9,10,11,12,13,14,15);
    const v16h b = __builtin_shufflevector(blo, bhi, 0,1,2,3,4,5,6,7,8,9,10,11,12,13,14,15);
    acc = __builtin_amdgcn_wmma_f32_16x16x32_f16(false, a, false, b,
                                                 (short)0, acc, false, false);
  }
  return acc;
}

// ---------------------------------------------------------------------------
// out(R,128) = [tanh(bias+)] fourier(kpos - qpos) @ W.  Row r -> (g, a, m),
// rel[c] = kpos[(g/kdiv)*Mk + m][c] - qpos[g*Aq + a][c].
// Fourier feature f = c*64 + s*32 + k : s==0 -> sin(rel*2^k), s==1 -> cos.
// ---------------------------------------------------------------------------
template <bool GEO>
__global__ void fourier_gemm_kernel(const float* __restrict__ kpos, int kstride,
                                    const float* __restrict__ qpos, int qstride,
                                    int Aq, int Mk, int kdiv,
                                    const f16* __restrict__ Wt,
                                    const float* __restrict__ bias,
                                    f16* __restrict__ out, int R) {
  __shared__ f16 sX[16 * LDSP];
  const int tid = threadIdx.x, lane = tid & 31, wave = tid >> 5;
  const int r0 = blockIdx.x * 16;
  for (int idx = tid; idx < 16 * 128; idx += 256) {
    const int row = idx >> 7, f = idx & 127;
    const int r = r0 + row;
    const int am = Aq * Mk;
    const int g = r / am, rem = r - g * am;
    const int a = rem / Mk, m = rem - a * Mk;
    const int c = f >> 6, s = (f >> 5) & 1, k = f & 31;
    float rel = kpos[((g / kdiv) * Mk + m) * kstride + c];
    if (qpos) rel -= qpos[(g * Aq + a) * qstride + c];
    const float ang = ldexpf(rel, k);
    const float val = s ? cosf(ang) : sinf(ang);
    sX[row * LDSP + f] = (f16)val;
  }
  __syncthreads();
  const int n0 = wave * 16;
  v8f acc = tile_gemm_128(sX, Wt, n0, lane);
  const int lrow = lane & 15, hs = lane >> 4, n = n0 + lrow;
  #pragma unroll
  for (int i = 0; i < 8; ++i) {
    const int r = r0 + i + 8 * hs;
    float v = acc[i];
    if (GEO) v = tanhf(v + bias[n]);
    out[r * 128 + n] = (f16)v;
  }
}

// out(R,128) = X(R,128) @ W(128,128); X f16, Wt f16 col-major, out f16.
__global__ void gemm128_kernel(const f16* __restrict__ X, const f16* __restrict__ Wt,
                               f16* __restrict__ out, int R) {
  __shared__ f16 sX[16 * LDSP];
  const int tid = threadIdx.x, lane = tid & 31, wave = tid >> 5;
  const int r0 = blockIdx.x * 16;
  {
    const int row = tid >> 4, coff = (tid & 15) << 3;   // one v8h chunk per thread
    *(v8h*)(sX + row * LDSP + coff) = *(const v8h*)(X + (r0 + row) * 128 + coff);
  }
  __syncthreads();
  const int n0 = wave * 16;
  v8f acc = tile_gemm_128(sX, Wt, n0, lane);
  const int lrow = lane & 15, hs = lane >> 4, n = n0 + lrow;
  #pragma unroll
  for (int i = 0; i < 8; ++i) {
    const int r = r0 + i + 8 * hs;
    out[r * 128 + n] = (f16)acc[i];
  }
}

// ---------------------------------------------------------------------------
// rho: 2 cross-attention layers per group. A=4, H=8, DH=16, M keys.
// ---------------------------------------------------------------------------
__global__ void rho_attn_kernel(const float* __restrict__ ainfo,        // (G,4,6)
                                const f16* __restrict__ kb0, const f16* __restrict__ vb0,
                                const f16* __restrict__ kb1, const f16* __restrict__ vb1, // (G,M,128)
                                const f16* __restrict__ e,               // (G,4,M,128)
                                const float* __restrict__ Win,           // (6,128)
                                const float* __restrict__ Wq, const float* __restrict__ Wo,
                                const float* __restrict__ W1, const float* __restrict__ W2, // (2,128,128)
                                float* __restrict__ out32, f16* __restrict__ out16, int M) {
  __shared__ float h[4 * 128];
  __shared__ float qb[4 * 128];
  __shared__ float ob[4 * 128];
  __shared__ float sc[4 * 8 * 256];
  const int g = blockIdx.x, tid = threadIdx.x;
  for (int idx = tid; idx < 512; idx += 256) {
    const int a = idx >> 7, d = idx & 127;
    float s = 0.f;
    #pragma unroll
    for (int i = 0; i < 6; ++i) s += ainfo[(g * 4 + a) * 6 + i] * Win[i * 128 + d];
    h[idx] = s;
  }
  __syncthreads();
  for (int l = 0; l < 2; ++l) {
    const f16* kb = l ? kb1 : kb0;
    const f16* vb = l ? vb1 : vb0;
    const float* wq = Wq + l * 16384; const float* wo = Wo + l * 16384;
    const float* w1 = W1 + l * 16384; const float* w2 = W2 + l * 16384;
    for (int idx = tid; idx < 512; idx += 256) {        // q = h @ Wq
      const int a = idx >> 7, d = idx & 127; float s = 0.f;
      for (int k = 0; k < 128; ++k) s += h[a * 128 + k] * wq[k * 128 + d];
      qb[idx] = s;
    }
    __syncthreads();
    for (int idx = tid; idx < 4 * 8 * M; idx += 256) {  // scores
      const int a = idx / (8 * M), hh = (idx / M) & 7, m = idx % M;
      const f16* kp = kb + (g * M + m) * 128 + hh * 16;
      const f16* ep = e + ((g * 4 + a) * M + m) * 128 + hh * 16;
      float s = 0.f;
      #pragma unroll
      for (int d = 0; d < 16; ++d)
        s += qb[a * 128 + hh * 16 + d] * ((float)kp[d] + (float)ep[d]);
      sc[(a * 8 + hh) * M + m] = s * 0.25f;
    }
    __syncthreads();
    if (tid < 32) {                                     // softmax over m
      float* row = sc + tid * M;
      float mx = -INFINITY;
      for (int m = 0; m < M; ++m) mx = fmaxf(mx, row[m]);
      float sum = 0.f;
      for (int m = 0; m < M; ++m) { const float ex = expf(row[m] - mx); row[m] = ex; sum += ex; }
      const float inv = 1.f / sum;
      for (int m = 0; m < M; ++m) row[m] *= inv;
    }
    __syncthreads();
    for (int idx = tid; idx < 512; idx += 256) {        // o = att @ (vb + e)
      const int a = idx >> 7, d = idx & 127, hh = d >> 4;
      const float* att = sc + (a * 8 + hh) * M;
      float s = 0.f;
      for (int m = 0; m < M; ++m)
        s += att[m] * ((float)vb[(g * M + m) * 128 + d] + (float)e[((g * 4 + a) * M + m) * 128 + d]);
      ob[idx] = s;
    }
    __syncthreads();
    for (int idx = tid; idx < 512; idx += 256) {        // h += o @ Wo
      const int a = idx >> 7, d = idx & 127; float s = 0.f;
      for (int k = 0; k < 128; ++k) s += ob[a * 128 + k] * wo[k * 128 + d];
      h[idx] += s;
    }
    __syncthreads();
    for (int idx = tid; idx < 512; idx += 256) {        // t = relu(h @ W1)
      const int a = idx >> 7, d = idx & 127; float s = 0.f;
      for (int k = 0; k < 128; ++k) s += h[a * 128 + k] * w1[k * 128 + d];
      qb[idx] = fmaxf(s, 0.f);
    }
    __syncthreads();
    for (int idx = tid; idx < 512; idx += 256) {        // h += t @ W2
      const int a = idx >> 7, d = idx & 127; float s = 0.f;
      for (int k = 0; k < 128; ++k) s += qb[a * 128 + k] * w2[k * 128 + d];
      h[idx] += s;
    }
    __syncthreads();
  }
  for (int idx = tid; idx < 512; idx += 256) {
    out32[g * 512 + idx] = h[idx];
    if (out16) out16[g * 512 + idx] = (f16)h[idx];
  }
}

// ---------------------------------------------------------------------------
// fuse: single cross-attention layer. Aq=4, Mk keys (<=80), kbase group = g/kdiv.
// ---------------------------------------------------------------------------
__global__ void fuse_attn_kernel(const float* __restrict__ qemb,   // (G,4,128)
                                 const f16* __restrict__ kb, const f16* __restrict__ vb, // (Gk,Mk,128)
                                 const f16* __restrict__ e,        // (G,4,Mk,128)
                                 const float* __restrict__ Wq, const float* __restrict__ Wo,
                                 const float* __restrict__ W1, const float* __restrict__ W2,
                                 float* __restrict__ out32, f16* __restrict__ out16,
                                 int Mk, int kdiv) {
  __shared__ float h[4 * 128];
  __shared__ float qb[4 * 128];
  __shared__ float ob[4 * 128];
  __shared__ float sc[4 * 8 * 80];
  const int g = blockIdx.x, tid = threadIdx.x;
  const int gk = g / kdiv;
  for (int idx = tid; idx < 512; idx += 256) h[idx] = qemb[g * 512 + idx];
  __syncthreads();
  for (int idx = tid; idx < 512; idx += 256) {
    const int a = idx >> 7, d = idx & 127; float s = 0.f;
    for (int k = 0; k < 128; ++k) s += h[a * 128 + k] * Wq[k * 128 + d];
    qb[idx] = s;
  }
  __syncthreads();
  for (int idx = tid; idx < 4 * 8 * Mk; idx += 256) {
    const int a = idx / (8 * Mk), hh = (idx / Mk) & 7, m = idx % Mk;
    const f16* kp = kb + (gk * Mk + m) * 128 + hh * 16;
    const f16* ep = e + ((g * 4 + a) * Mk + m) * 128 + hh * 16;
    float s = 0.f;
    #pragma unroll
    for (int d = 0; d < 16; ++d)
      s += qb[a * 128 + hh * 16 + d] * ((float)kp[d] + (float)ep[d]);
    sc[(a * 8 + hh) * Mk + m] = s * 0.25f;
  }
  __syncthreads();
  if (tid < 32) {
    float* row = sc + tid * Mk;
    float mx = -INFINITY;
    for (int m = 0; m < Mk; ++m) mx = fmaxf(mx, row[m]);
    float sum = 0.f;
    for (int m = 0; m < Mk; ++m) { const float ex = expf(row[m] - mx); row[m] = ex; sum += ex; }
    const float inv = 1.f / sum;
    for (int m = 0; m < Mk; ++m) row[m] *= inv;
  }
  __syncthreads();
  for (int idx = tid; idx < 512; idx += 256) {
    const int a = idx >> 7, d = idx & 127, hh = d >> 4;
    const float* att = sc + (a * 8 + hh) * Mk;
    float s = 0.f;
    for (int m = 0; m < Mk; ++m)
      s += att[m] * ((float)vb[(gk * Mk + m) * 128 + d] + (float)e[((g * 4 + a) * Mk + m) * 128 + d]);
    ob[idx] = s;
  }
  __syncthreads();
  for (int idx = tid; idx < 512; idx += 256) {
    const int a = idx >> 7, d = idx & 127; float s = 0.f;
    for (int k = 0; k < 128; ++k) s += ob[a * 128 + k] * Wo[k * 128 + d];
    h[idx] += s;
  }
  __syncthreads();
  for (int idx = tid; idx < 512; idx += 256) {
    const int a = idx >> 7, d = idx & 127; float s = 0.f;
    for (int k = 0; k < 128; ++k) s += h[a * 128 + k] * W1[k * 128 + d];
    qb[idx] = fmaxf(s, 0.f);
  }
  __syncthreads();
  for (int idx = tid; idx < 512; idx += 256) {
    const int a = idx >> 7, d = idx & 127; float s = 0.f;
    for (int k = 0; k < 128; ++k) s += qb[a * 128 + k] * W2[k * 128 + d];
    const float v = h[idx] + s;
    out32[g * 512 + idx] = v;
    if (out16) out16[g * 512 + idx] = (f16)v;
  }
}

// reverse_action scan: block per b.
__global__ void reverse_action_kernel(const float* __restrict__ ainfo,   // (8,4,6)
                                      const float* __restrict__ opos,    // (8,256,2)
                                      const float* __restrict__ actions, // (8,8,4)
                                      float* __restrict__ pred_obj,      // (64,256,2)
                                      float* __restrict__ pred_agent) {  // (64,4,6)
  const int b = blockIdx.x, tid = threadIdx.x;
  const int T = 8, M = 256, A = 4;
  if (tid < M) {
    float px = opos[(b * M + tid) * 2 + 0], py = opos[(b * M + tid) * 2 + 1];
    for (int t = 0; t < T; ++t) {
      const float* act = actions + (b * T + t) * 4;
      const float x = px - act[0], y = py - act[1];
      const float c = cosf(-act[2]), s = sinf(-act[2]);
      px = c * x + s * y;
      py = -s * x + c * y;
      pred_obj[((b * T + t) * M + tid) * 2 + 0] = px;
      pred_obj[((b * T + t) * M + tid) * 2 + 1] = py;
    }
  }
  if (tid < A) {
    const float* ai = ainfo + (b * A + tid) * 6;
    float grip = ai[4];
    for (int t = 0; t < T; ++t) {
      float sa = actions[(b * T + t) * 4 + 3];
      sa = fminf(fmaxf(sa, 0.f), 1.f);
      if (rintf(sa) != rintf(grip)) grip = sa;
      float* pa = pred_agent + ((b * T + t) * A + tid) * 6;
      pa[0] = ai[0]; pa[1] = ai[1]; pa[2] = ai[2]; pa[3] = ai[3]; pa[4] = grip; pa[5] = ai[5];
    }
  }
}

// head: pooled = mean_a final; out = relu(pooled@W1+b1)@W2+b2.  64 threads/block.
__global__ void head_kernel(const float* __restrict__ fin,  // (64,4,128)
                            const float* __restrict__ W1, const float* __restrict__ b1,
                            const float* __restrict__ W2, const float* __restrict__ b2,
                            float* __restrict__ out) {
  __shared__ float pooled[128];
  __shared__ float hid[64];
  const int bt = blockIdx.x, tid = threadIdx.x;
  for (int d = tid; d < 128; d += 64) {
    float s = 0.f;
    #pragma unroll
    for (int a = 0; a < 4; ++a) s += fin[(bt * 4 + a) * 128 + d];
    pooled[d] = s * 0.25f;
  }
  __syncthreads();
  {
    float s = b1[tid];
    for (int d = 0; d < 128; ++d) s += pooled[d] * W1[d * 64 + tid];
    hid[tid] = fmaxf(s, 0.f);
  }
  __syncthreads();
  if (tid < 4) {
    float s = b2[tid];
    for (int j = 0; j < 64; ++j) s += hid[j] * W2[j * 4 + tid];
    out[bt * 4 + tid] = s;
  }
}

// ---------------------------------------------------------------------------
extern "C" void kernel_launch(void* const* d_in, const int* in_sizes, int n_in,
                              void* d_out, int out_size, void* d_ws, size_t ws_size,
                              hipStream_t stream) {
  (void)in_sizes; (void)n_in; (void)out_size; (void)ws_size;
  const float* curr_ai = (const float*)d_in[0];   // (8,4,6)
  const float* curr_op = (const float*)d_in[1];   // (8,256,2)
  const float* demo_ai = (const float*)d_in[2];   // (8,2,10,4,6) = (160,4,6)
  const float* demo_op = (const float*)d_in[3];   // (8,2,10,256,2) = (160,256,2)
  const float* actions = (const float*)d_in[4];   // (8,8,4)
  const float* geW   = (const float*)d_in[5];
  const float* geB   = (const float*)d_in[6];
  const float* rWin  = (const float*)d_in[7];
  const float* rWe   = (const float*)d_in[8];
  const float* rWq   = (const float*)d_in[9];
  const float* rWk   = (const float*)d_in[10];
  const float* rWv   = (const float*)d_in[11];
  const float* rWo   = (const float*)d_in[12];
  const float* rW1   = (const float*)d_in[13];
  const float* rW2   = (const float*)d_in[14];
  const float* pWq   = (const float*)d_in[15];
  const float* pWk   = (const float*)d_in[16];
  const float* pWv   = (const float*)d_in[17];
  const float* pWo   = (const float*)d_in[18];
  const float* pWe   = (const float*)d_in[19];
  const float* pW1   = (const float*)d_in[20];
  const float* pW2   = (const float*)d_in[21];
  const float* sWq   = (const float*)d_in[22];
  const float* sWk   = (const float*)d_in[23];
  const float* sWv   = (const float*)d_in[24];
  const float* sWo   = (const float*)d_in[25];
  const float* sWe   = (const float*)d_in[26];
  const float* sW1   = (const float*)d_in[27];
  const float* sW2   = (const float*)d_in[28];
  const float* hW1   = (const float*)d_in[29];
  const float* hB1   = (const float*)d_in[30];
  const float* hW2   = (const float*)d_in[31];
  const float* hB2   = (const float*)d_in[32];

  size_t off = 0;
  auto a16 = [&](size_t n) { f16* p = (f16*)((char*)d_ws + off); off += ((n * 2 + 255) / 256) * 256; return p; };
  auto a32 = [&](size_t n) { float* p = (float*)((char*)d_ws + off); off += ((n * 4 + 255) / 256) * 256; return p; };

  // transposed f16 weights for WMMA GEMMs (col-major: Wt[n][k])
  f16* tGe  = a16(16384);
  f16* tRe  = a16(16384);
  f16* tRk0 = a16(16384); f16* tRk1 = a16(16384);
  f16* tRv0 = a16(16384); f16* tRv1 = a16(16384);
  f16* tPe  = a16(16384); f16* tPk  = a16(16384); f16* tPv = a16(16384);
  f16* tSe  = a16(16384); f16* tSk  = a16(16384); f16* tSv = a16(16384);

  // workspace
  f16* demoF = a16((size_t)160 * 256 * 128);
  f16* eDemo = a16((size_t)160 * 4 * 256 * 128);
  f16* kd0 = a16((size_t)160 * 256 * 128); f16* vd0 = a16((size_t)160 * 256 * 128);
  f16* kd1 = a16((size_t)160 * 256 * 128); f16* vd1 = a16((size_t)160 * 256 * 128);
  float* demoRho32 = a32((size_t)160 * 4 * 128); f16* demoRho16 = a16((size_t)160 * 4 * 128);
  f16* currF = a16((size_t)8 * 256 * 128);
  f16* eCurr = a16((size_t)8 * 4 * 256 * 128);
  f16* kc0 = a16((size_t)8 * 256 * 128); f16* vc0 = a16((size_t)8 * 256 * 128);
  f16* kc1 = a16((size_t)8 * 256 * 128); f16* vc1 = a16((size_t)8 * 256 * 128);
  float* currRho32 = a32((size_t)8 * 4 * 128);
  float* predObj = a32((size_t)64 * 256 * 2);
  float* predAg  = a32((size_t)64 * 4 * 6);
  f16* predF = a16((size_t)64 * 256 * 128);
  f16* ePred = a16((size_t)64 * 4 * 256 * 128);
  f16* kp0 = a16((size_t)64 * 256 * 128); f16* vp0 = a16((size_t)64 * 256 * 128);
  f16* kp1 = a16((size_t)64 * 256 * 128); f16* vp1 = a16((size_t)64 * 256 * 128);
  float* predRho32 = a32((size_t)64 * 4 * 128);
  f16* ePhi = a16((size_t)8 * 4 * 80 * 128);
  f16* kbPhi = a16((size_t)8 * 80 * 128); f16* vbPhi = a16((size_t)8 * 80 * 128);
  float* ctx32 = a32((size_t)8 * 4 * 128); f16* ctx16 = a16((size_t)8 * 4 * 128);
  f16* ePsi = a16((size_t)64 * 4 * 4 * 128);
  f16* ctxK = a16((size_t)8 * 4 * 128); f16* ctxV = a16((size_t)8 * 4 * 128);
  float* fin = a32((size_t)64 * 4 * 128);

  // ---- weight transposes (once per launch) ----
  transpose_w_kernel<<<64, 256, 0, stream>>>(geW, tGe);
  transpose_w_kernel<<<64, 256, 0, stream>>>(rWe, tRe);
  transpose_w_kernel<<<64, 256, 0, stream>>>(rWk, tRk0);
  transpose_w_kernel<<<64, 256, 0, stream>>>(rWk + 16384, tRk1);
  transpose_w_kernel<<<64, 256, 0, stream>>>(rWv, tRv0);
  transpose_w_kernel<<<64, 256, 0, stream>>>(rWv + 16384, tRv1);
  transpose_w_kernel<<<64, 256, 0, stream>>>(pWe, tPe);
  transpose_w_kernel<<<64, 256, 0, stream>>>(pWk, tPk);
  transpose_w_kernel<<<64, 256, 0, stream>>>(pWv, tPv);
  transpose_w_kernel<<<64, 256, 0, stream>>>(sWe, tSe);
  transpose_w_kernel<<<64, 256, 0, stream>>>(sWk, tSk);
  transpose_w_kernel<<<64, 256, 0, stream>>>(sWv, tSv);

  // ---- demo path ----
  int R;
  R = 160 * 256;   // demo geo: F = tanh(fourier(P)@geW + geB)
  fourier_gemm_kernel<true><<<R / 16, 256, 0, stream>>>(demo_op, 2, nullptr, 0, 1, 256, 1, tGe, geB, demoF, R);
  R = 160 * 4 * 256;  // e = fourier(scene - agent_xy) @ rho_We
  fourier_gemm_kernel<false><<<R / 16, 256, 0, stream>>>(demo_op, 2, demo_ai, 6, 4, 256, 1, tRe, nullptr, eDemo, R);
  R = 160 * 256;
  gemm128_kernel<<<R / 16, 256, 0, stream>>>(demoF, tRk0, kd0, R);
  gemm128_kernel<<<R / 16, 256, 0, stream>>>(demoF, tRv0, vd0, R);
  gemm128_kernel<<<R / 16, 256, 0, stream>>>(demoF, tRk1, kd1, R);
  gemm128_kernel<<<R / 16, 256, 0, stream>>>(demoF, tRv1, vd1, R);
  rho_attn_kernel<<<160, 256, 0, stream>>>(demo_ai, kd0, vd0, kd1, vd1, eDemo,
                                           rWin, rWq, rWo, rW1, rW2, demoRho32, demoRho16, 256);

  // ---- curr path ----
  R = 8 * 256;
  fourier_gemm_kernel<true><<<R / 16, 256, 0, stream>>>(curr_op, 2, nullptr, 0, 1, 256, 1, tGe, geB, currF, R);
  R = 8 * 4 * 256;
  fourier_gemm_kernel<false><<<R / 16, 256, 0, stream>>>(curr_op, 2, curr_ai, 6, 4, 256, 1, tRe, nullptr, eCurr, R);
  R = 8 * 256;
  gemm128_kernel<<<R / 16, 256, 0, stream>>>(currF, tRk0, kc0, R);
  gemm128_kernel<<<R / 16, 256, 0, stream>>>(currF, tRv0, vc0, R);
  gemm128_kernel<<<R / 16, 256, 0, stream>>>(currF, tRk1, kc1, R);
  gemm128_kernel<<<R / 16, 256, 0, stream>>>(currF, tRv1, vc1, R);
  rho_attn_kernel<<<8, 256, 0, stream>>>(curr_ai, kc0, vc0, kc1, vc1, eCurr,
                                         rWin, rWq, rWo, rW1, rW2, currRho32, nullptr, 256);

  // ---- reverse action ----
  reverse_action_kernel<<<8, 256, 0, stream>>>(curr_ai, curr_op, actions, predObj, predAg);

  // ---- pred path ----
  R = 64 * 256;
  fourier_gemm_kernel<true><<<R / 16, 256, 0, stream>>>(predObj, 2, nullptr, 0, 1, 256, 1, tGe, geB, predF, R);
  R = 64 * 4 * 256;
  fourier_gemm_kernel<false><<<R / 16, 256, 0, stream>>>(predObj, 2, predAg, 6, 4, 256, 1, tRe, nullptr, ePred, R);
  R = 64 * 256;
  gemm128_kernel<<<R / 16, 256, 0, stream>>>(predF, tRk0, kp0, R);
  gemm128_kernel<<<R / 16, 256, 0, stream>>>(predF, tRv0, vp0, R);
  gemm128_kernel<<<R / 16, 256, 0, stream>>>(predF, tRk1, kp1, R);
  gemm128_kernel<<<R / 16, 256, 0, stream>>>(predF, tRv1, vp1, R);
  rho_attn_kernel<<<64, 256, 0, stream>>>(predAg, kp0, vp0, kp1, vp1, ePred,
                                          rWin, rWq, rWo, rW1, rW2, predRho32, nullptr, 256);

  // ---- phi fuse: ctx = fuse(curr_rho, demo_rho) ----
  R = 8 * 4 * 80;
  fourier_gemm_kernel<false><<<R / 16, 256, 0, stream>>>(demo_ai, 6, curr_ai, 6, 4, 80, 1, tPe, nullptr, ePhi, R);
  R = 8 * 80;
  gemm128_kernel<<<R / 16, 256, 0, stream>>>(demoRho16, tPk, kbPhi, R);
  gemm128_kernel<<<R / 16, 256, 0, stream>>>(demoRho16, tPv, vbPhi, R);
  fuse_attn_kernel<<<8, 256, 0, stream>>>(currRho32, kbPhi, vbPhi, ePhi,
                                          pWq, pWo, pW1, pW2, ctx32, ctx16, 80, 1);

  // ---- psi fuse: final = fuse(pred_rho, ctx broadcast over T) ----
  R = 64 * 4 * 4;
  fourier_gemm_kernel<false><<<R / 16, 256, 0, stream>>>(curr_ai, 6, predAg, 6, 4, 4, 8, tSe, nullptr, ePsi, R);
  R = 8 * 4;
  gemm128_kernel<<<R / 16, 256, 0, stream>>>(ctx16, tSk, ctxK, R);
  gemm128_kernel<<<R / 16, 256, 0, stream>>>(ctx16, tSv, ctxV, R);
  fuse_attn_kernel<<<64, 256, 0, stream>>>(predRho32, ctxK, ctxV, ePsi,
                                           sWq, sWo, sW1, sW2, fin, nullptr, 4, 8);

  // ---- head ----
  head_kernel<<<64, 64, 0, stream>>>(fin, hW1, hB1, hW2, hB2, (float*)d_out);
}
